// AttentionGNN_64647847739971
// MI455X (gfx1250) — matrix operand order repaired and checked by reading, not compiled, not run on hardware
//
#include <hip/hip_runtime.h>
#include <hip/hip_fp16.h>

#define NODES   50000
#define EDGES   800000
#define ETOT    (EDGES + NODES)
#define IN_DIM  128
#define HID     32
#define HEADS   8
#define F1      (HEADS * HID)   /* 256 */
#define OUT_DIM 16
#define NEG_SLOPE 0.2f

typedef __attribute__((ext_vector_type(16))) _Float16 v16h;
typedef __attribute__((ext_vector_type(8)))  _Float16 v8h;
typedef __attribute__((ext_vector_type(8)))  float    v8f;

// ---------- float <-> order-preserving uint (for atomic max on floats) ------
__device__ __forceinline__ unsigned enc_f(float x) {
    unsigned u = __float_as_uint(x);
    return (u & 0x80000000u) ? ~u : (u | 0x80000000u);
}
__device__ __forceinline__ float dec_f(unsigned u) {
    return (u & 0x80000000u) ? __uint_as_float(u & 0x7FFFFFFFu)
                             : __uint_as_float(~u);
}
__device__ __forceinline__ float lrelu(float x) {
    return x > 0.f ? x : NEG_SLOPE * x;
}

// ---------------------------------------------------------------------------
// fp32 -> fp16 row-major copy (A operands: x, h1, h2)
// ---------------------------------------------------------------------------
__global__ void cvt_f16(const float* __restrict__ in, _Float16* __restrict__ out,
                        int n) {
    const int i = blockIdx.x * blockDim.x + threadIdx.x;
    if (i < n) out[i] = (_Float16)in[i];
}

// ---------------------------------------------------------------------------
// fp32 [K,N] -> fp16 [N,K] transpose (weight operands, tiny)
// ---------------------------------------------------------------------------
__global__ void cvt_tr_f16(const float* __restrict__ in, _Float16* __restrict__ out,
                           int N, int K) {
    const int i = blockIdx.x * blockDim.x + threadIdx.x;
    if (i < N * K) {
        const int n = i / K, k = i % K;
        out[i] = (_Float16)in[(size_t)k * N + n];
    }
}

// ---------------------------------------------------------------------------
// Mixed-precision GEMM: C[M,N] = A[M,K] @ B[K,N] (+bias), f32 accumulate.
// A: fp16 [M,K] row-major.  BT: fp16 [N,K] (pre-transposed weights).
// One wave per 16x16 C tile; all waves in a block share one A tile, staged
// once in LDS (row stride K+8 halfs -> conflict-free b128 lane reads).
// Launch geometry must cover tiles exactly (no early-exit waves: barriers +
// WMMA EXEC=all-ones stay valid). M multiple of 16, K multiple of 32.
// ---------------------------------------------------------------------------
template <int K, bool BIAS>
__global__ void gat_gemm_wmma(const _Float16* __restrict__ A,
                              const _Float16* __restrict__ BT,
                              float* __restrict__ C, int N,
                              const float* __restrict__ bias) {
    constexpr int LDA = K + 8;                  // pad: lanes hit banks 0,4,..,60
    __shared__ _Float16 As[16 * LDA];

    const int lane = threadIdx.x & 31;
    const int wave = threadIdx.x >> 5;
    const int mt   = blockIdx.x;
    const int nt   = blockIdx.y * (blockDim.x >> 5) + wave;
    const int row0 = mt * 16, col0 = nt * 16;
    const int hi = lane >> 4;                   // half-wave select
    const int mn = lane & 15;                   // M index (A) / N index (B,C)

    // Cooperative A-tile stage: tile = contiguous 16*K halfs at A[row0*K].
    const _Float16* Atile = A + (size_t)row0 * K;
    constexpr int CH = 16 * K / 8;              // number of 8-half chunks
    for (int c = threadIdx.x; c < CH; c += blockDim.x) {
        const int r = c / (K / 8), off = (c % (K / 8)) * 8;
        *(v8h*)(As + r * LDA + off) = *(const v8h*)(Atile + r * K + off);
    }
    __syncthreads();

    v8f acc = {};
    const _Float16* Bbase = BT + (size_t)(col0 + mn) * K;
    #pragma unroll
    for (int k0 = 0; k0 < K; k0 += 32) {
        // A 16x32 f16 frag: per lane, K in [8*hi, 8*hi+7] and [16+8*hi, ..+7]
        const _Float16* Ar = As + mn * LDA + k0 + 8 * hi;
        const v8h alo = *(const v8h*)(Ar);
        const v8h ahi = *(const v8h*)(Ar + 16);
        // B 32x16 f16 frag: per lane, K in [8*hi, 8*hi+7] and [16+8*hi, ..+7]
        const _Float16* Br = Bbase + k0 + 8 * hi;
        const v8h blo = *(const v8h*)(Br);
        const v8h bhi = *(const v8h*)(Br + 16);
        const v16h a = __builtin_shufflevector(alo, ahi, 0, 1, 2, 3, 4, 5, 6, 7,
                                               8, 9, 10, 11, 12, 13, 14, 15);
        const v16h b = __builtin_shufflevector(blo, bhi, 0, 1, 2, 3, 4, 5, 6, 7,
                                               8, 9, 10, 11, 12, 13, 14, 15);
        acc = __builtin_amdgcn_wmma_f32_16x16x32_f16(
                  false, a, false, b, (short)0, acc, false, false);
    }

    const float bv = BIAS ? bias[col0 + mn] : 0.f;
    float* Crow = C + (size_t)(row0 + 8 * hi) * N + col0 + mn;
    #pragma unroll
    for (int r = 0; r < 8; ++r)                  // C/D: V_r -> M=r+8*hi
        Crow[(size_t)r * N] = acc[r] + bv;
}

// ---------------------------------------------------------------------------
// Per-(node,head) attention scalars: as[i] = <h[i,:], a_src[h]>, likewise ad.
// ---------------------------------------------------------------------------
__global__ void gat_attn_coeffs(const float* __restrict__ h,
                                const float* __restrict__ a_src,
                                const float* __restrict__ a_dst,
                                float* __restrict__ as, float* __restrict__ ad,
                                int nH, int D) {
    const int i = blockIdx.x * blockDim.x + threadIdx.x;
    if (i >= NODES * nH) return;
    const float* hp = h + (size_t)i * D;
    const float* sv = a_src + (i % nH) * D;
    const float* dv = a_dst + (i % nH) * D;
    float accs = 0.f, accd = 0.f;
    for (int d = 0; d < D; ++d) { float v = hp[d]; accs += v * sv[d]; accd += v * dv[d]; }
    as[i] = accs; ad[i] = accd;
}

// ---------------------------------------------------------------------------
// Pass 1: segment max of leaky_relu(as[src]+ad[dst]) per dst node.
// ---------------------------------------------------------------------------
__global__ void gat_edge_max(const int* __restrict__ ei,
                             const float* __restrict__ as,
                             const float* __restrict__ ad,
                             unsigned* __restrict__ m, int nH) {
    const int e = blockIdx.x * blockDim.x + threadIdx.x;
    if (e >= ETOT) return;
    const int s = (e < EDGES) ? ei[e] : (e - EDGES);
    const int d = (e < EDGES) ? ei[EDGES + e] : (e - EDGES);
    for (int hh = 0; hh < nH; ++hh) {
        const float x = lrelu(as[s * nH + hh] + ad[d * nH + hh]);
        atomicMax(m + d * nH + hh, enc_f(x));
    }
}

// ---------------------------------------------------------------------------
// Pass 2: den[dst] += exp(logit - max[dst])
// ---------------------------------------------------------------------------
__global__ void gat_edge_expsum(const int* __restrict__ ei,
                                const float* __restrict__ as,
                                const float* __restrict__ ad,
                                const unsigned* __restrict__ m,
                                float* __restrict__ den, int nH) {
    const int e = blockIdx.x * blockDim.x + threadIdx.x;
    if (e >= ETOT) return;
    const int s = (e < EDGES) ? ei[e] : (e - EDGES);
    const int d = (e < EDGES) ? ei[EDGES + e] : (e - EDGES);
    for (int hh = 0; hh < nH; ++hh) {
        const float x = lrelu(as[s * nH + hh] + ad[d * nH + hh]);
        atomicAdd(den + d * nH + hh, expf(x - dec_f(m[d * nH + hh])));
    }
}

// ---------------------------------------------------------------------------
// Pass 3: out[dst,h,:] += alpha * h[src,h,:]   (one thread per (edge,head))
// ---------------------------------------------------------------------------
__global__ void gat_edge_agg(const int* __restrict__ ei,
                             const float* __restrict__ as,
                             const float* __restrict__ ad,
                             const unsigned* __restrict__ m,
                             const float* __restrict__ den,
                             const float* __restrict__ h,
                             float* __restrict__ out, int nH, int D) {
    const int t = blockIdx.x * blockDim.x + threadIdx.x;
    if (t >= ETOT * nH) return;
    const int e = t / nH, hh = t % nH;
    const int s = (e < EDGES) ? ei[e] : (e - EDGES);
    const int d = (e < EDGES) ? ei[EDGES + e] : (e - EDGES);
    const float x = lrelu(as[s * nH + hh] + ad[d * nH + hh]);
    const float alpha = expf(x - dec_f(m[d * nH + hh])) /
                        (den[d * nH + hh] + 1e-16f);
    const float* hs = h + ((size_t)s * nH + hh) * D;
    float* od = out + ((size_t)d * nH + hh) * D;
    for (int dd = 0; dd < D; ++dd) atomicAdd(od + dd, alpha * hs[dd]);
}

// ---------------------------------------------------------------------------
// In-place ELU(x + bias)
// ---------------------------------------------------------------------------
__global__ void gat_elu_bias(float* __restrict__ x, const float* __restrict__ b,
                             int Cdim) {
    const int i = blockIdx.x * blockDim.x + threadIdx.x;
    if (i >= NODES * Cdim) return;
    const float v = x[i] + b[i % Cdim];
    x[i] = v > 0.f ? v : expm1f(v);
}

// ---------------------------------------------------------------------------
extern "C" void kernel_launch(void* const* d_in, const int* in_sizes, int n_in,
                              void* d_out, int out_size, void* d_ws, size_t ws_size,
                              hipStream_t stream) {
    const float* x      = (const float*)d_in[0];
    const int*   ei     = (const int*)  d_in[1];
    const float* W1     = (const float*)d_in[2];
    const float* a_src1 = (const float*)d_in[3];
    const float* a_dst1 = (const float*)d_in[4];
    const float* b1     = (const float*)d_in[5];
    const float* W2     = (const float*)d_in[6];
    const float* a_src2 = (const float*)d_in[7];
    const float* a_dst2 = (const float*)d_in[8];
    const float* b2     = (const float*)d_in[9];
    const float* Wout   = (const float*)d_in[10];
    const float* bout   = (const float*)d_in[11];
    float* out = (float*)d_out;

    // Workspace layout. Layer-2 f32 scratch reuses the h1p region; the f16
    // A-operand buffer `ah` is reused for x, h1, h2 in turn.
    float*    ws   = (float*)d_ws;
    float*    h1p  = ws;                                   // [N,256] f32
    float*    agg1 = h1p + (size_t)NODES * F1;             // [N,256] -> h1 (ELU)
    float*    as1  = agg1 + (size_t)NODES * F1;            // [N,8]
    float*    ad1  = as1 + NODES * HEADS;                  // [N,8]
    unsigned* m1   = (unsigned*)(ad1 + NODES * HEADS);     // [N,8]
    float*    den1 = (float*)(m1 + NODES * HEADS);         // [N,8]
    _Float16* ah   = (_Float16*)(den1 + NODES * HEADS);    // [N,256] f16 (shared)
    _Float16* w1t  = ah + (size_t)NODES * F1;              // [256,128] f16
    _Float16* w2t  = w1t + F1 * IN_DIM;                    // [32,256]  f16
    _Float16* wot  = w2t + HID * F1;                       // [16,32]   f16

    float*    h2p  = h1p;                                  // [N,32] f32
    float*    agg2 = h1p + (size_t)NODES * HID;            // [N,32] -> h2 (ELU)
    float*    as2  = agg2 + (size_t)NODES * HID;           // [N]
    float*    ad2  = as2 + NODES;                          // [N]
    unsigned* m2   = (unsigned*)(ad2 + NODES);             // [N]
    float*    den2 = (float*)(m2 + NODES);                 // [N]

    const int EB = (ETOT + 255) / 256;

    // ---- One-time operand prep -------------------------------------------
    cvt_tr_f16<<<(F1 * IN_DIM + 255) / 256, 256, 0, stream>>>(W1, w1t, F1, IN_DIM);
    cvt_tr_f16<<<(HID * F1 + 255) / 256, 256, 0, stream>>>(W2, w2t, HID, F1);
    cvt_tr_f16<<<(OUT_DIM * HID + 255) / 256, 256, 0, stream>>>(Wout, wot, OUT_DIM, HID);

    // ---- Layer 1: GATConv(128 -> 32, heads=8) -----------------------------
    hipMemsetAsync(agg1, 0, (size_t)NODES * F1 * sizeof(float), stream);
    hipMemsetAsync(m1,   0, (size_t)NODES * HEADS * sizeof(float) * 2, stream); // m1+den1

    cvt_f16<<<(NODES * IN_DIM + 255) / 256, 256, 0, stream>>>(x, ah, NODES * IN_DIM);
    gat_gemm_wmma<IN_DIM, false><<<dim3(NODES / 16, 2), 256, 0, stream>>>(
        ah, w1t, h1p, F1, nullptr);
    gat_attn_coeffs<<<(NODES * HEADS + 255) / 256, 256, 0, stream>>>(
        h1p, a_src1, a_dst1, as1, ad1, HEADS, HID);
    gat_edge_max<<<EB, 256, 0, stream>>>(ei, as1, ad1, m1, HEADS);
    gat_edge_expsum<<<EB, 256, 0, stream>>>(ei, as1, ad1, m1, den1, HEADS);
    gat_edge_agg<<<(ETOT * HEADS + 255) / 256, 256, 0, stream>>>(
        ei, as1, ad1, m1, den1, h1p, agg1, HEADS, HID);
    gat_elu_bias<<<(NODES * F1 + 255) / 256, 256, 0, stream>>>(agg1, b1, F1);

    // ---- Layer 2: GATConv(256 -> 32, heads=1) -----------------------------
    cvt_f16<<<(NODES * F1 + 255) / 256, 256, 0, stream>>>(agg1, ah, NODES * F1);
    gat_gemm_wmma<F1, false><<<dim3(NODES / 16, 1), 64, 0, stream>>>(
        ah, w2t, h2p, HID, nullptr);
    hipMemsetAsync(agg2, 0, (size_t)NODES * HID * sizeof(float), stream);
    hipMemsetAsync(m2,   0, (size_t)NODES * sizeof(float) * 2, stream);        // m2+den2

    gat_attn_coeffs<<<(NODES + 255) / 256, 256, 0, stream>>>(
        h2p, a_src2, a_dst2, as2, ad2, 1, HID);
    gat_edge_max<<<EB, 256, 0, stream>>>(ei, as2, ad2, m2, 1);
    gat_edge_expsum<<<EB, 256, 0, stream>>>(ei, as2, ad2, m2, den2, 1);
    gat_edge_agg<<<EB, 256, 0, stream>>>(
        ei, as2, ad2, m2, den2, h2p, agg2, 1, HID);
    gat_elu_bias<<<(NODES * HID + 255) / 256, 256, 0, stream>>>(agg2, b2, HID);

    // ---- Output head: out = h2 @ Wout + bout ------------------------------
    cvt_f16<<<(NODES * HID + 255) / 256, 256, 0, stream>>>(agg2, ah, NODES * HID);
    gat_gemm_wmma<HID, true><<<dim3(NODES / 16, 1), 32, 0, stream>>>(
        ah, wot, out, OUT_DIM, bout);
}
